// DepthWiseSepConv_36945308680637
// MI455X (gfx1250) — matrix operand order repaired and checked by reading, not compiled, not run on hardware
//
#include <hip/hip_runtime.h>
#include <stddef.h>
#include <stdint.h>

// ---------------------------------------------------------------------------
// Types for CDNA5 WMMA / TDM
// ---------------------------------------------------------------------------
typedef __attribute__((ext_vector_type(16))) __bf16 v16bf;
typedef __attribute__((ext_vector_type(8)))  __bf16 v8bf;
typedef __attribute__((ext_vector_type(8)))  float  v8f;
typedef unsigned int u32x4 __attribute__((ext_vector_type(4)));
typedef int          i32x4 __attribute__((ext_vector_type(4)));
typedef int          i32x8 __attribute__((ext_vector_type(8)));

#define EPS 1e-5f
#define B_  64
#define C_  240
#define HW_ 784        // 28*28
#define R_  60
#define O_  240
#define KPAD 256       // K=240 padded to 8 WMMA k-steps of 32
#define TILE_P 64      // pixels per GEMM block

__device__ __forceinline__ float hswish(float v) {
    float c = fminf(fmaxf(v + 3.0f, 0.0f), 6.0f);
    return v * c * (1.0f / 6.0f);
}

// fp32 -> bf16 round-to-nearest-even
__device__ __forceinline__ unsigned short f2bf(float f) {
    unsigned u = __builtin_bit_cast(unsigned, f);
    unsigned r = u + 0x7FFFu + ((u >> 16) & 1u);
    return (unsigned short)(r >> 16);
}

// ---------------------------------------------------------------------------
// Kernel 1: depthwise 5x5 + bias + BN1 + hardswish -> bf16 dwT[b][p][cpad]
//           (+ SE global-average-pool sums). c-contiguous layout so the
//           GEMM kernel can TDM-load [pixel][K] tiles directly.
// grid = (C, B), block = 256
// ---------------------------------------------------------------------------
__global__ __launch_bounds__(256) void dw_hs_kernel(
    const float* __restrict__ x, const float* __restrict__ dw_w,
    const float* __restrict__ dw_b,
    const float* __restrict__ g1, const float* __restrict__ b1,
    const float* __restrict__ m1, const float* __restrict__ v1,
    unsigned short* __restrict__ dwT, float* __restrict__ sbuf)
{
    __shared__ float tile[32 * 32];   // 28x28 plane + halo of 2
    __shared__ float wsh[25];
    __shared__ float red[256];

    const int c   = blockIdx.x;
    const int b   = blockIdx.y;
    const int tid = threadIdx.x;

    if (tid < 25) wsh[tid] = dw_w[c * 25 + tid];

    const float* xc = x + ((size_t)b * C_ + c) * HW_;
#pragma unroll
    for (int it = 0; it < 4; ++it) {
        int i  = tid + it * 256;          // 32x32 padded tile
        int y  = i >> 5, xx = i & 31;
        int gy = y - 2,  gx = xx - 2;
        float v = 0.0f;
        if (gy >= 0 && gy < 28 && gx >= 0 && gx < 28) v = xc[gy * 28 + gx];
        tile[i] = v;
    }
    __syncthreads();

    const float scale = g1[c] * rsqrtf(v1[c] + EPS);
    const float shift = b1[c] - m1[c] * scale;
    const float bias  = dw_b[c];

    float localSum = 0.0f;
#pragma unroll
    for (int it = 0; it < 4; ++it) {
        int p = tid + it * 256;
        if (p < HW_) {
            int oy = p / 28;
            int ox = p - oy * 28;
            float a = 0.0f;
#pragma unroll
            for (int ky = 0; ky < 5; ++ky)
#pragma unroll
                for (int kx = 0; kx < 5; ++kx)
                    a += wsh[ky * 5 + kx] * tile[(oy + ky) * 32 + (ox + kx)];
            a = (a + bias) * scale + shift;
            a = hswish(a);
            localSum += a;
            dwT[((size_t)b * HW_ + p) * KPAD + c] = f2bf(a);
        }
    }

    // channel blocks c==0 also zero the K-pad columns [240,256)
    if (c == 0) {
        const uint4 z = {0u, 0u, 0u, 0u};
#pragma unroll
        for (int it = 0; it < 4; ++it) {
            int p = tid + it * 256;
            if (p < HW_) {
                uint4* dst = (uint4*)&dwT[((size_t)b * HW_ + p) * KPAD + C_];
                dst[0] = z;   // 16 bf16 = 32 B
                dst[1] = z;
            }
        }
    }

    // block reduction for SE pooling
    red[tid] = localSum;
    __syncthreads();
#pragma unroll
    for (int s = 128; s > 0; s >>= 1) {
        if (tid < s) red[tid] += red[tid + s];
        __syncthreads();
    }
    if (tid == 0) sbuf[b * C_ + c] = red[0] * (1.0f / (float)HW_);
}

// ---------------------------------------------------------------------------
// Kernel 2: pw_w [O,C] fp32 -> bf16 [O, KPAD] (zero-padded K)
// ---------------------------------------------------------------------------
__global__ __launch_bounds__(256) void pw_cvt_kernel(
    const float* __restrict__ pw_w, unsigned short* __restrict__ pwbf)
{
    const int o = blockIdx.x, c = threadIdx.x;
    pwbf[o * KPAD + c] = (c < C_) ? f2bf(pw_w[o * C_ + c]) : (unsigned short)0;
}

// ---------------------------------------------------------------------------
// Kernel 3: SE MLP  g = hardswish(relu(s@W1^T + b1) @ W2^T + b2)
// ---------------------------------------------------------------------------
__global__ __launch_bounds__(256) void se_kernel(
    const float* __restrict__ sbuf,
    const float* __restrict__ w1, const float* __restrict__ bb1,
    const float* __restrict__ w2, const float* __restrict__ bb2,
    float* __restrict__ gbuf)
{
    __shared__ float sh_s[C_];
    __shared__ float sh_h[R_];
    const int b = blockIdx.x, t = threadIdx.x;

    if (t < C_) sh_s[t] = sbuf[b * C_ + t];
    __syncthreads();
    if (t < R_) {
        float acc = bb1[t];
        for (int c = 0; c < C_; ++c) acc += w1[t * C_ + c] * sh_s[c];
        sh_h[t] = fmaxf(acc, 0.0f);
    }
    __syncthreads();
    if (t < O_) {
        float acc = bb2[t];
#pragma unroll
        for (int r = 0; r < R_; ++r) acc += w2[t * R_ + r] * sh_h[r];
        gbuf[b * O_ + t] = hswish(acc);
    }
}

// ---------------------------------------------------------------------------
// Kernel 4: pointwise GEMM. M = 64 pixels (TDM-staged in LDS), N = 240 out
// channels, K = 256 (padded). 8 waves = 4 pixel M-tiles x 2 channel groups.
//   - activation tile moved Global->LDS by the Tensor Data Mover
//   - A fragments: contiguous LDS b128 reads (contiguous k = 16*half + s order)
//   - B fragments: contiguous 32B global reads of bf16 weights (L2-resident)
//   - epilogue: bias + SE gate + BN2, two float4 stores per lane
// grid = (ceil(784/64)=13, B), block = 256
// ---------------------------------------------------------------------------
__global__ __launch_bounds__(256) void pw_gemm_kernel(
    const unsigned short* __restrict__ dwT,
    const unsigned short* __restrict__ pwbf,
    const float* __restrict__ pw_b, const float* __restrict__ gbuf,
    const float* __restrict__ g2, const float* __restrict__ b2,
    const float* __restrict__ m2, const float* __restrict__ v2,
    float* __restrict__ out)
{
    __shared__ unsigned short ldsA[TILE_P * KPAD];   // [pixel][K] bf16, 32 KB

    const int b   = blockIdx.y;
    const int p0  = blockIdx.x * TILE_P;
    const int tid = threadIdx.x;

    // ---- TDM: DMA the [TILE_P x KPAD] bf16 tile into LDS (wave 0 issues) ---
    if (tid < 32) {
        const unsigned long long gaddr =
            (unsigned long long)(uintptr_t)(dwT + ((size_t)b * HW_ + p0) * KPAD);
        const unsigned lds_off  = (unsigned)(uintptr_t)(&ldsA[0]); // LDS aperture: low 32 bits = LDS byte offset
        const unsigned rows_rem = (unsigned)(HW_ - p0);            // OOB rows -> zero fill

        // D# group 0: count=1 | lds_addr | global_addr | type=2
        u32x4 g0 = { 1u,
                     lds_off,
                     (unsigned)(gaddr & 0xFFFFFFFFull),
                     (unsigned)((gaddr >> 32) & 0x01FFFFFFull) | 0x80000000u };
        // D# group 1: data_size=2B; tensor_dim0=KPAD; tensor_dim1=rows_rem;
        //             tile_dim0=KPAD; tile_dim1=TILE_P; dim0_stride=KPAD
        i32x8 g1 = { (int)(1u << 16),
                     (int)((unsigned)KPAD << 16),
                     (int)((rows_rem & 0xFFFFu) << 16),
                     (int)((unsigned)KPAD << 16),
                     (int)TILE_P,
                     (int)KPAD,
                     0, 0 };
        i32x4 gz = {0, 0, 0, 0};
#if __clang_major__ >= 23
        i32x8 gz8 = {0, 0, 0, 0, 0, 0, 0, 0};
        __builtin_amdgcn_tensor_load_to_lds(g0, g1, gz, gz, gz8, 0);
#else
        __builtin_amdgcn_tensor_load_to_lds(g0, g1, gz, gz, 0);
#endif
        __builtin_amdgcn_s_wait_tensorcnt(0);
    }
    __syncthreads();

    const int lane   = tid & 31;
    const int wave   = tid >> 5;
    const int lane16 = lane & 15;
    const int half   = lane >> 4;

    const int mrow    = (wave & 3) * 16 + lane16;      // pixel row in tile (A: M)
    const int ntBeg   = (wave >> 2) * 8;               // output-channel tiles
    const int ntEnd   = (ntBeg + 8 < 15) ? (ntBeg + 8) : 15;
    const int pixBase = p0 + (wave & 3) * 16 + half * 8;

    // preload all 8 A fragments: per lane two contiguous 16B LDS reads
    v16bf afrag[8];
#pragma unroll
    for (int kk = 0; kk < 8; ++kk) {
        const int c0 = kk * 32 + half * 16;
        const unsigned short* ap = &ldsA[mrow * KPAD + c0];
        const v8bf lo = *(const v8bf*)(ap);
        const v8bf hi = *(const v8bf*)(ap + 8);
        v16bf a;
#pragma unroll
        for (int i = 0; i < 8; ++i) { a[i] = lo[i]; a[i + 8] = hi[i]; }
        afrag[kk] = a;
    }

    for (int nt = ntBeg; nt < ntEnd; ++nt) {
        const int o = nt * 16 + lane16;                // out channel (B: N)
        v8f acc = {0.f, 0.f, 0.f, 0.f, 0.f, 0.f, 0.f, 0.f};
#pragma unroll
        for (int kk = 0; kk < 8; ++kk) {
            // B fragment: 16 contiguous bf16 from weight row (same k = 16*half + s order as A)
            const v16bf bfr =
                *(const v16bf*)(pwbf + o * KPAD + kk * 32 + half * 16);
            acc = __builtin_amdgcn_wmma_f32_16x16x32_bf16(
                      false, afrag[kk], false, bfr, (short)0, acc, false, false);
        }
        // epilogue: one channel per lane; 8 consecutive pixels per lane
        const float sc = g2[o] * rsqrtf(v2[o] + EPS);
        const float sh = b2[o] - m2[o] * sc;
        const float gg = gbuf[b * O_ + o];
        const float pb = pw_b[o];
        float* op = out + (size_t)b * (O_ * (size_t)HW_) + (size_t)o * HW_ + pixBase;
        if (pixBase + 7 < HW_) {
            float4 f0, f1;
            f0.x = fmaf((acc[0] + pb) * gg, sc, sh);
            f0.y = fmaf((acc[1] + pb) * gg, sc, sh);
            f0.z = fmaf((acc[2] + pb) * gg, sc, sh);
            f0.w = fmaf((acc[3] + pb) * gg, sc, sh);
            f1.x = fmaf((acc[4] + pb) * gg, sc, sh);
            f1.y = fmaf((acc[5] + pb) * gg, sc, sh);
            f1.z = fmaf((acc[6] + pb) * gg, sc, sh);
            f1.w = fmaf((acc[7] + pb) * gg, sc, sh);
            *(float4*)(op)     = f0;
            *(float4*)(op + 4) = f1;
        } else {
#pragma unroll
            for (int r = 0; r < 8; ++r)
                if (pixBase + r < HW_)
                    op[r] = fmaf((acc[r] + pb) * gg, sc, sh);
        }
    }
}

// ---------------------------------------------------------------------------
// Launch
// ---------------------------------------------------------------------------
extern "C" void kernel_launch(void* const* d_in, const int* in_sizes, int n_in,
                              void* d_out, int out_size, void* d_ws, size_t ws_size,
                              hipStream_t stream) {
    const float* x     = (const float*)d_in[0];
    const float* dw_w  = (const float*)d_in[1];
    const float* dw_b  = (const float*)d_in[2];
    const float* bn1_g = (const float*)d_in[3];
    const float* bn1_b = (const float*)d_in[4];
    const float* bn1_m = (const float*)d_in[5];
    const float* bn1_v = (const float*)d_in[6];
    const float* pw_w  = (const float*)d_in[7];
    const float* pw_b  = (const float*)d_in[8];
    const float* se_w1 = (const float*)d_in[9];
    const float* se_b1 = (const float*)d_in[10];
    const float* se_w2 = (const float*)d_in[11];
    const float* se_b2 = (const float*)d_in[12];
    const float* bn2_g = (const float*)d_in[13];
    const float* bn2_b = (const float*)d_in[14];
    const float* bn2_m = (const float*)d_in[15];
    const float* bn2_v = (const float*)d_in[16];
    float* out = (float*)d_out;

    // workspace partition
    char* ws = (char*)d_ws;
    unsigned short* dwT  = (unsigned short*)ws;                 // 25,690,112 B
    float* sbuf          = (float*)(ws + 25690112);             //    245,760 B
    float* gbuf          = (float*)(ws + 25935872);             //    245,760 B
    unsigned short* pwbf = (unsigned short*)(ws + 26181632);    //    122,880 B

    dw_hs_kernel<<<dim3(C_, B_), 256, 0, stream>>>(
        x, dw_w, dw_b, bn1_g, bn1_b, bn1_m, bn1_v, dwT, sbuf);

    pw_cvt_kernel<<<dim3(O_), 256, 0, stream>>>(pw_w, pwbf);

    se_kernel<<<dim3(B_), 256, 0, stream>>>(
        sbuf, se_w1, se_b1, se_w2, se_b2, gbuf);

    pw_gemm_kernel<<<dim3((HW_ + TILE_P - 1) / TILE_P, B_), 256, 0, stream>>>(
        dwT, pwbf, pw_b, gbuf, bn2_g, bn2_b, bn2_m, bn2_v, out);
}
